// Agent_29755533427644
// MI455X (gfx1250) — matrix operand order repaired and checked by reading, not compiled, not run on hardware
//
#include <hip/hip_runtime.h>

typedef __attribute__((ext_vector_type(16))) _Float16 v16h;
typedef __attribute__((ext_vector_type(8)))  _Float16 v8h;
typedef __attribute__((ext_vector_type(8)))  float    v8f;

// ---------------- problem / tiling constants ----------------
constexpr int kMaxN = 20;          // nodes per graph
constexpr int kHid  = 256;
constexpr int kG    = 4;           // graphs per workgroup
constexpr int kRows = kG * kMaxN;  // 80 rows -> exactly 5 M-tiles of 16
constexpr int kMT   = kRows / 16;  // 5
constexpr int kSA   = 520;         // abuf row stride (halfs), 16B aligned, bank-skewed
constexpr int kSH   = 260;         // hbuf row stride (floats), bank-skewed
constexpr int kSQ   = 264;         // qa row stride (halfs)
constexpr int kNW   = 8;           // waves per block (256 threads)
constexpr float kEps = 1e-5f;

// ---------------- f16 transposed-weight workspace offsets (in halfs) --------
constexpr int OFF_NE   = 0;                   // neW^T  [256][32]  (K=16 padded to 32)
constexpr int OFF_G1W1 = OFF_NE   + 256*32;   // g1W1^T [256][512]
constexpr int OFF_G1W2 = OFF_G1W1 + 256*512;  // g1W2^T [256][256]
constexpr int OFF_G2W1 = OFF_G1W2 + 256*256;  // g2W1^T [256][512]
constexpr int OFF_G2W2 = OFF_G2W1 + 256*512;  // g2W2^T [256][256]
constexpr int OFF_PQ   = OFF_G2W2 + 256*256;  // pqW^T  [256][256]
constexpr int OFF_PK   = OFF_PQ   + 256*256;  // pkW^T  [256][256]

// ---------------- dynamic LDS layout (bytes) ----------------
constexpr int L_HBUF  = 0;                      // 80*260*4 = 83200  (f32 activations)
constexpr int L_ABUF  = L_HBUF + kRows*kSH*4;   // 80*520*2 = 83200  (f16 GEMM A matrix)
constexpr int L_GBUF  = L_ABUF + kRows*kSA*2;   // 4*256*4           (graph embedding g)
constexpr int L_QBUF  = L_GBUF + kG*kHid*4;     // 4*256*4           (q vectors)
constexpr int L_QA    = L_QBUF + kG*kHid*4;     // 16*264*2          (padded A for q GEMM)
constexpr int L_MSK   = L_QA   + 16*kSQ*2;      // 80*4              (row mask)
constexpr int L_INV   = L_MSK  + kRows*4;       // 4*4               (1/clip(count,1))
constexpr int L_TOTAL = L_INV  + kG*4;          // ~183 KB

// ---------------- WMMA helpers ----------------
__device__ __forceinline__ v8f wmma_f16(v16h a, v16h b, v8f c) {
  return __builtin_amdgcn_wmma_f32_16x16x32_f16(false, a, false, b, (short)0, c,
                                                false, false);
}

// A fragment (16x32 f16) from LDS, row-major with stride (halfs).
// ISA layout: lane m=L&15, grp=L>>4; halfs 0..7 -> K = k0+grp*8+{0..7},
// halfs 8..15 -> K = k0+16+grp*8+{0..7}.  Two ds_load_b128 per lane.
__device__ __forceinline__ v16h ldsA(const _Float16* abase, int lane, int mtile,
                                     int k0, int stride) {
  int m = mtile * 16 + (lane & 15);
  int grp = lane >> 4;
  const _Float16* p = abase + m * stride + k0 + grp * 8;
  union { v16h v; v8h h[2]; } u;
  u.h[0] = *(const v8h*)(p);
  u.h[1] = *(const v8h*)(p + 16);
  return u.v;
}

// B fragment (32x16 f16) from transposed weights W^T[N][ks] in global.
// ISA layout: lane n=L&15 is the column, grp=L>>4; halfs 0..7 -> K = k0+grp*8+{0..7},
// halfs 8..15 -> K = k0+16+grp*8+{0..7} (same order as ldsA).
__device__ __forceinline__ v16h gB(const _Float16* wt, int lane, int n0, int k0,
                                   int ks) {
  int n = n0 + (lane & 15);
  int kb = k0 + (lane >> 4) * 8;
  const _Float16* p = wt + n * ks + kb;
  union { v16h v; v8h h[2]; } u;
  u.h[0] = *(const v8h*)(p);
  u.h[1] = *(const v8h*)(p + 16);
  return u.v;
}

// Full [80 x KTOT] @ [KTOT x 32] K-loop for one wave's two N-tiles.
template <int KTOT>
__device__ __forceinline__ void gemm_big(v8f (&acc)[kMT][2], const _Float16* abase,
                                         const _Float16* wt, int ks, int lane,
                                         int n0) {
  v8f z = {};
#pragma unroll
  for (int mt = 0; mt < kMT; ++mt) { acc[mt][0] = z; acc[mt][1] = z; }
  for (int k = 0; k < KTOT; k += 32) {
    v16h b0 = gB(wt, lane, n0, k, ks);
    v16h b1 = gB(wt, lane, n0 + 16, k, ks);
#pragma unroll
    for (int mt = 0; mt < kMT; ++mt) {
      v16h a = ldsA(abase, lane, mt, k, kSA);
      acc[mt][0] = wmma_f16(a, b0, acc[mt][0]);
      acc[mt][1] = wmma_f16(a, b1, acc[mt][1]);
    }
  }
}

// masked mean over nodes -> broadcast f16 into abuf cols [256..511]
__device__ __forceinline__ void msg_broadcast(const float* hbuf, _Float16* abuf,
                                              const float* maskf, const float* invc,
                                              int t) {
  for (int idx = t; idx < kG * kHid; idx += 256) {
    int g = idx >> 8, c = idx & 255;
    float s = 0.f;
#pragma unroll
    for (int n = 0; n < kMaxN; ++n)
      s += hbuf[(g * kMaxN + n) * kSH + c] * maskf[g * kMaxN + n];
    s *= invc[g];
    _Float16 hf = (_Float16)s;
#pragma unroll
    for (int n = 0; n < kMaxN; ++n)
      abuf[(g * kMaxN + n) * kSA + kHid + c] = hf;
  }
}

// row-wise LayerNorm * mask; writes f32 back to hbuf and f16 into abuf[0..255]
__device__ __forceinline__ void layernorm_mask(float* hbuf, _Float16* abuf,
                                               const float* lg, const float* lb,
                                               const float* maskf, int wav, int lane) {
#pragma unroll
  for (int rr = 0; rr < kRows / kNW; ++rr) {
    int row = wav * (kRows / kNW) + rr;
    float x[8];
    float s = 0.f;
#pragma unroll
    for (int i = 0; i < 8; ++i) { x[i] = hbuf[row * kSH + lane + i * 32]; s += x[i]; }
    for (int off = 16; off; off >>= 1) s += __shfl_xor(s, off, 32);
    float mean = s * (1.0f / kHid);
    float s2 = 0.f;
#pragma unroll
    for (int i = 0; i < 8; ++i) { float d = x[i] - mean; s2 += d * d; }
    for (int off = 16; off; off >>= 1) s2 += __shfl_xor(s2, off, 32);
    float rstd = rsqrtf(s2 * (1.0f / kHid) + kEps);
    float mk = maskf[row];
#pragma unroll
    for (int i = 0; i < 8; ++i) {
      int c = lane + i * 32;
      float v = ((x[i] - mean) * rstd * lg[c] + lb[c]) * mk;
      hbuf[row * kSH + c] = v;
      abuf[row * kSA + c] = (_Float16)v;
    }
  }
}

// ---------------- weight prep: f32 [K][N] -> f16 [N][Kpad] (zero-pad K) -------
__global__ void wprep_kernel(const float* __restrict__ src, _Float16* __restrict__ dst,
                             int K, int N, int Kpad) {
  int idx = blockIdx.x * blockDim.x + threadIdx.x;
  if (idx >= N * Kpad) return;
  int n = idx / Kpad, k = idx - n * Kpad;
  dst[idx] = (k < K) ? (_Float16)src[(size_t)k * N + n] : (_Float16)0.f;
}

// ---------------- fused GNN kernel: 4 graphs per workgroup ----------------
__global__ void __launch_bounds__(256) gnn_fused(
    const float* __restrict__ nf, const int* __restrict__ nn_c,
    const _Float16* __restrict__ ws,
    const float* __restrict__ neb,
    const float* __restrict__ g1b1, const float* __restrict__ g1b2,
    const float* __restrict__ g2b1, const float* __restrict__ g2b2,
    const float* __restrict__ n1g, const float* __restrict__ n1b,
    const float* __restrict__ n2g, const float* __restrict__ n2b,
    const float* __restrict__ ohW, const float* __restrict__ ohb,
    const float* __restrict__ c1W, const float* __restrict__ c1b,
    const float* __restrict__ c2W, const float* __restrict__ c2b,
    const float* __restrict__ pqb, const float* __restrict__ pkb,
    float* __restrict__ out_oh, float* __restrict__ out_c1,
    float* __restrict__ out_c2, float* __restrict__ out_pl) {
  extern __shared__ char smem[];
  float*    hbuf  = (float*)(smem + L_HBUF);
  _Float16* abuf  = (_Float16*)(smem + L_ABUF);
  float*    gbuf  = (float*)(smem + L_GBUF);
  float*    qbuf  = (float*)(smem + L_QBUF);
  _Float16* qa    = (_Float16*)(smem + L_QA);
  float*    maskf = (float*)(smem + L_MSK);
  float*    invc  = (float*)(smem + L_INV);

  const int t    = threadIdx.x;
  const int lane = t & 31;
  const int wav  = t >> 5;
  const int n0   = wav * 32;            // this wave's 32-column slice
  const int colA = n0 + (lane & 15);
  const int grp  = lane >> 4;
  const int b0   = blockIdx.x * kG;

  // ---- masks / denominators ----
  if (t < kG) {
    int c = nn_c[b0 + t];
    if (c < 0) c = 0;
    if (c > kMaxN) c = kMaxN;
    invc[t] = 1.f / (float)(c < 1 ? 1 : c);
  }
  if (t < kRows) {
    int g = t / kMaxN, n = t - g * kMaxN;
    maskf[t] = (n < nn_c[b0 + g]) ? 1.f : 0.f;
  }

  // ---- stage node features as f16 A0 [80][32] (K padded 16->32) ----
  const float* nfp = nf + (size_t)b0 * kMaxN * 16;
  for (int idx = t; idx < kRows * 16; idx += 256) {
    int r = idx >> 4, f = idx & 15;
    abuf[r * kSA + f]      = (_Float16)nfp[idx];
    abuf[r * kSA + 16 + f] = (_Float16)0.f;
  }
  __syncthreads();

  v8f acc[kMT][2];

  // ---- encoder: h = nf @ neW + neb  (K=32, single WMMA step) ----
  {
    v16h bA = gB(ws + OFF_NE, lane, n0, 0, 32);
    v16h bB = gB(ws + OFF_NE, lane, n0 + 16, 0, 32);
    v8f z = {};
#pragma unroll
    for (int mt = 0; mt < kMT; ++mt) {
      v16h a = ldsA(abuf, lane, mt, 0, kSA);
      acc[mt][0] = wmma_f16(a, bA, z);
      acc[mt][1] = wmma_f16(a, bB, z);
    }
    __syncthreads();  // all A0 reads done before overwriting abuf
    float bs0 = neb[colA], bs1 = neb[colA + 16];
#pragma unroll
    for (int mt = 0; mt < kMT; ++mt)
#pragma unroll
      for (int r = 0; r < 8; ++r) {
        int row = mt * 16 + r + grp * 8;
        float v0 = acc[mt][0][r] + bs0;
        float v1 = acc[mt][1][r] + bs1;
        hbuf[row * kSH + colA]      = v0;
        hbuf[row * kSH + colA + 16] = v1;
        abuf[row * kSA + colA]      = (_Float16)v0;
        abuf[row * kSA + colA + 16] = (_Float16)v1;
      }
    __syncthreads();
  }

  // ================= two message-passing blocks =================
#pragma unroll 1
  for (int blk = 0; blk < 2; ++blk) {
    const _Float16* w1 = ws + (blk == 0 ? OFF_G1W1 : OFF_G2W1);
    const _Float16* w2 = ws + (blk == 0 ? OFF_G1W2 : OFF_G2W2);
    const float* bb1 = (blk == 0) ? g1b1 : g2b1;
    const float* bb2 = (blk == 0) ? g1b2 : g2b2;
    const float* lg  = (blk == 0) ? n1g : n2g;
    const float* lb  = (blk == 0) ? n1b : n2b;

    msg_broadcast(hbuf, abuf, maskf, invc, t);   // fills abuf cols 256..511
    __syncthreads();

    // layer 1: t = relu([h|msg] @ W1 + b1)   (K=512)
    gemm_big<512>(acc, abuf, w1, 512, lane, n0);
    __syncthreads();                             // everyone done reading msg cols
    {
      float b1a = bb1[colA], b1b = bb1[colA + 16];
#pragma unroll
      for (int mt = 0; mt < kMT; ++mt)
#pragma unroll
        for (int r = 0; r < 8; ++r) {
          int row = mt * 16 + r + grp * 8;
          float v0 = acc[mt][0][r] + b1a; v0 = v0 > 0.f ? v0 : 0.f;
          float v1 = acc[mt][1][r] + b1b; v1 = v1 > 0.f ? v1 : 0.f;
          abuf[row * kSA + kHid + colA]      = (_Float16)v0;  // reuse msg cols for t
          abuf[row * kSA + kHid + colA + 16] = (_Float16)v1;
        }
    }
    __syncthreads();

    // layer 2 + residual: h = h + (t @ W2 + b2)   (K=256, A at abuf cols 256..511)
    gemm_big<256>(acc, abuf + kHid, w2, 256, lane, n0);
    {
      float b2a = bb2[colA], b2b = bb2[colA + 16];
#pragma unroll
      for (int mt = 0; mt < kMT; ++mt)
#pragma unroll
        for (int r = 0; r < 8; ++r) {
          int row = mt * 16 + r + grp * 8;
          hbuf[row * kSH + colA]      += acc[mt][0][r] + b2a;  // wave owns these entries
          hbuf[row * kSH + colA + 16] += acc[mt][1][r] + b2b;
        }
    }
    __syncthreads();

    layernorm_mask(hbuf, abuf, lg, lb, maskf, wav, lane);
    __syncthreads();
  }

  // ---- graph embedding g = masked mean; build padded A for q GEMM ----
  for (int idx = t; idx < kG * kHid; idx += 256) {
    int g = idx >> 8, c = idx & 255;
    float s = 0.f;
#pragma unroll
    for (int n = 0; n < kMaxN; ++n) s += hbuf[(g * kMaxN + n) * kSH + c];
    s *= invc[g];                       // h already masked by LN*mf
    gbuf[g * kHid + c] = s;
    qa[g * kSQ + c] = (_Float16)s;
  }
  for (int idx = t; idx < (16 - kG) * kHid; idx += 256) {
    int g = kG + idx / kHid, c = idx % kHid;
    qa[g * kSQ + c] = (_Float16)0.f;
  }
  __syncthreads();

  // ---- q = g @ pqW + pqb  (1 M-tile WMMA) ----
  {
    v8f q0 = {}, q1 = {};
    for (int k = 0; k < kHid; k += 32) {
      v16h a = ldsA(qa, lane, 0, k, kSQ);
      q0 = wmma_f16(a, gB(ws + OFF_PQ, lane, n0, k, 256), q0);
      q1 = wmma_f16(a, gB(ws + OFF_PQ, lane, n0 + 16, k, 256), q1);
    }
    if (grp == 0) {
#pragma unroll
      for (int r = 0; r < kG; ++r) {
        qbuf[r * kHid + colA]      = q0[r] + pqb[colA];
        qbuf[r * kHid + colA + 16] = q1[r] + pqb[colA + 16];
      }
    }
  }

  // ---- k = h @ pkW + pkb  (WMMA), overwrite hbuf with k ----
  gemm_big<256>(acc, abuf, ws + OFF_PK, 256, lane, n0);
  {
    float pk0 = pkb[colA], pk1 = pkb[colA + 16];
#pragma unroll
    for (int mt = 0; mt < kMT; ++mt)
#pragma unroll
      for (int r = 0; r < 8; ++r) {
        int row = mt * 16 + r + grp * 8;
        hbuf[row * kSH + colA]      = acc[mt][0][r] + pk0;
        hbuf[row * kSH + colA + 16] = acc[mt][1][r] + pk1;
      }
  }
  __syncthreads();

  // ---- tiny heads: oh/c1/c2 (VALU dots from gbuf) ----
  if (t < kG * 28) {
    int g = t / 28, o = t - g * 28;
    const float* gv = gbuf + g * kHid;
    int b = b0 + g;
    if (o < 8) {
      float s = ohb[o];
      for (int c = 0; c < kHid; ++c) s += gv[c] * ohW[c * 8 + o];
      out_oh[(size_t)b * 8 + o] = s;
    } else if (o < 18) {
      int j = o - 8;
      float s = c1b[j];
      for (int c = 0; c < kHid; ++c) s += gv[c] * c1W[c * 10 + j];
      out_c1[(size_t)b * 10 + j] = s;
    } else {
      int j = o - 18;
      float s = c2b[j];
      for (int c = 0; c < kHid; ++c) s += gv[c] * c2W[c * 10 + j];
      out_c2[(size_t)b * 10 + j] = s;
    }
  }

  // ---- pl[b,n] = q[b] . k[b,n]  (wave32 shuffle reduction), masked ----
#pragma unroll
  for (int rr = 0; rr < kRows / kNW; ++rr) {
    int row = wav * (kRows / kNW) + rr;
    int g = row / kMaxN, n = row - g * kMaxN;
    float s = 0.f;
#pragma unroll
    for (int i = 0; i < 8; ++i) {
      int c = lane + i * 32;
      s += qbuf[g * kHid + c] * hbuf[row * kSH + c];
    }
    for (int off = 16; off; off >>= 1) s += __shfl_xor(s, off, 32);
    if (lane == 0)
      out_pl[(size_t)(b0 + g) * kMaxN + n] = (maskf[row] > 0.f) ? s : -1e9f;
  }
}

// ---------------- launch ----------------
extern "C" void kernel_launch(void* const* d_in, const int* in_sizes, int n_in,
                              void* d_out, int out_size, void* d_ws, size_t ws_size,
                              hipStream_t stream) {
  const int Btot = in_sizes[1];          // nn_c has B elements
  _Float16* ws = (_Float16*)d_ws;

  auto prep = [&](int i, int off, int K, int N, int Kpad) {
    int total = N * Kpad;
    wprep_kernel<<<(total + 255) / 256, 256, 0, stream>>>(
        (const float*)d_in[i], ws + off, K, N, Kpad);
  };
  prep(2,  OFF_NE,   16,  256, 32);
  prep(4,  OFF_G1W1, 512, 256, 512);
  prep(6,  OFF_G1W2, 256, 256, 256);
  prep(8,  OFF_G2W1, 512, 256, 512);
  prep(10, OFF_G2W2, 256, 256, 256);
  prep(22, OFF_PQ,   256, 256, 256);
  prep(24, OFF_PK,   256, 256, 256);

  float* out    = (float*)d_out;
  float* out_oh = out;
  float* out_c1 = out + (size_t)Btot * 8;
  float* out_c2 = out + (size_t)Btot * 18;
  float* out_pl = out + (size_t)Btot * 28;

  gnn_fused<<<Btot / kG, 32 * kNW, L_TOTAL, stream>>>(
      (const float*)d_in[0], (const int*)d_in[1], ws,
      (const float*)d_in[3],
      (const float*)d_in[5], (const float*)d_in[7],
      (const float*)d_in[9], (const float*)d_in[11],
      (const float*)d_in[12], (const float*)d_in[13],
      (const float*)d_in[14], (const float*)d_in[15],
      (const float*)d_in[16], (const float*)d_in[17],
      (const float*)d_in[18], (const float*)d_in[19],
      (const float*)d_in[20], (const float*)d_in[21],
      (const float*)d_in[23], (const float*)d_in[25],
      out_oh, out_c1, out_c2, out_pl);
}
